// PartEmbedder_85512798863889
// MI455X (gfx1250) — matrix operand order repaired and checked by reading, not compiled, not run on hardware
//
#include <hip/hip_runtime.h>
#include <hip/hip_bf16.h>

typedef unsigned short     u16;
typedef unsigned int       u32;
typedef unsigned long long u64;
typedef __attribute__((ext_vector_type(16))) __bf16 v16bf;
typedef __attribute__((ext_vector_type(8)))  float  v8f;
typedef __attribute__((ext_vector_type(4)))  unsigned int v4u;
typedef __attribute__((ext_vector_type(4)))  int    v4i;
typedef __attribute__((ext_vector_type(8)))  int    v8i;

#define NN   65536
#define DD   512
#define EE   524288
#define BB   64
#define LL   3

union Frag { v16bf v; uint4 q[2]; };

__device__ __forceinline__ u16 f2bf(float f) {
    u32 u = __float_as_uint(f);
    u32 r = u + 0x7FFFu + ((u >> 16) & 1u);
    return (u16)(r >> 16);
}

// ---------------------------------------------------------------------------
// Tensor Data Mover: 2D tile load Global->LDS, D# built per ISA 8.3/8.4.
//   group0: count=1 | lds_addr | global_addr[56:0] | type=2
//   group1: data_size=1 (2B), tensor_dim0/1 (OOB rows read as zero),
//           tile_dim0/1, tensor_dim0_stride
// ---------------------------------------------------------------------------
__device__ __forceinline__ void tdm_load_2d(u32 lds_off, const void* gptr,
                                            u32 tile_d0, u32 tile_d1,
                                            u32 ten_d0, u32 ten_d1, u32 stride0) {
    u64 ga = (u64)(size_t)gptr;
    v4u g0 = { 1u,                                   // count=1 (valid), no gather
               lds_off,                              // LDS byte address
               (u32)ga,                              // global_addr[31:0]
               ((u32)(ga >> 32) & 0x01FFFFFFu) | 0x80000000u };  // ga[56:32] | type=2
    v8i g1;
    g1[0] = (int)(1u << 16);                         // workgroup_mask=0, data_size=2B
    g1[1] = (int)((ten_d0 & 0xFFFFu) << 16);         // tensor_dim0[15:0] @ bits 63:48
    g1[2] = (int)((ten_d0 >> 16) | ((ten_d1 & 0xFFFFu) << 16));   // dim0 hi | dim1 lo
    g1[3] = (int)((ten_d1 >> 16) | ((tile_d0 & 0xFFFFu) << 16));  // dim1 hi | tile_dim0
    g1[4] = (int)(tile_d1 & 0xFFFFu);                // tile_dim1 (0 = 1D), tile_dim2=0
    g1[5] = (int)stride0;                            // tensor_dim0_stride[31:0]
    g1[6] = 0;                                       // stride hi, dim1_stride lo
    g1[7] = 0;
    v4i z4 = {0, 0, 0, 0};
#if __has_include(<hip/amd_detail/amd_gfx1250_TDM.h>)
    v8i z8 = {0, 0, 0, 0, 0, 0, 0, 0};
    __builtin_amdgcn_tensor_load_to_lds(g0, g1, z4, z4, z8, 0);
#else
    __builtin_amdgcn_tensor_load_to_lds(g0, g1, z4, z4, 0);
#endif
}

// ---------------------------------------------------------------------------
// Pack fp32 weights -> bf16 in WMMA-B fragment order.
// pack[mat][kt][nt][lane][j]: lane l holds col n = nt*16+(l&15),
// K = kt*32 + (l>>4)*16 + j  => frag = 32B contiguous per lane.
// mats: 0=proj, 1..3=lin1[i], 4..6=lin2[i], 7..9=pool[i]
// ---------------------------------------------------------------------------
__global__ void pack_weights_kernel(const float* __restrict__ proj_w,
                                    const float* __restrict__ lin1_w,
                                    const float* __restrict__ lin2_w,
                                    const float* __restrict__ pool_w,
                                    u16* __restrict__ pack) {
    int gid = blockIdx.x * 256 + threadIdx.x;      // 10*16*32*32 = 163840
    int mat = gid >> 14;
    int idx = gid & 16383;
    int kt = idx >> 10;
    int nt = (idx >> 5) & 31;
    int l  = idx & 31;
    const float* src;
    if (mat == 0)      src = proj_w;
    else if (mat < 4)  src = lin1_w + (size_t)(mat - 1) * DD * DD;
    else if (mat < 7)  src = lin2_w + (size_t)(mat - 4) * DD * DD;
    else               src = pool_w + (size_t)(mat - 7) * DD * DD;
    int n  = nt * 16 + (l & 15);
    int kb = kt * 32 + (l >> 4) * 16;
    u16* dst = pack + (size_t)mat * DD * DD + ((size_t)(kt * 32 + nt) * 32 + l) * 16;
#pragma unroll
    for (int j = 0; j < 16; ++j)
        dst[j] = f2bf(src[(size_t)(kb + j) * DD + n]);
}

// ---------------------------------------------------------------------------
// Segment max via monotonic-uint atomicMax (native global_atomic_max_u32)
// ---------------------------------------------------------------------------
__global__ void segmax_kernel(const float* __restrict__ x,
                              const int* __restrict__ batch,
                              u32* __restrict__ keys) {
    int n = blockIdx.x;
    int b = batch[n];
    const float* xr = x + (size_t)n * DD;
    u32* kb = keys + (size_t)b * DD;
    for (int c = threadIdx.x; c < DD; c += blockDim.x) {
        u32 u = __float_as_uint(xr[c]);
        u32 key = ((int)u >= 0) ? (u | 0x80000000u) : ~u;
        atomicMax(&kb[c], key);
    }
}

__global__ void decode_segmax_kernel(const u32* __restrict__ keys,
                                     u16* __restrict__ outbf) {
    int i = blockIdx.x * 256 + threadIdx.x;        // B*D
    u32 key = keys[i];
    u32 u = (key & 0x80000000u) ? (key ^ 0x80000000u) : ~key;
    outbf[i] = f2bf(__uint_as_float(u));
}

// ---------------------------------------------------------------------------
// Edge scatter-add: one wave per edge, coalesced lanes, HW f32 atomics
// ---------------------------------------------------------------------------
__global__ void scatter_add_kernel(const float* __restrict__ x,
                                   const int* __restrict__ ei,
                                   float* __restrict__ agg) {
    int e    = blockIdx.x * 8 + (threadIdx.x >> 5);
    int lane = threadIdx.x & 31;
    int s = ei[e];
    int d = ei[EE + e];
    const float* xs = x + (size_t)s * DD;
    float* ad = agg + (size_t)d * DD;
#pragma unroll
    for (int j = 0; j < 16; ++j) {
        int c = j * 32 + lane;
        unsafeAtomicAdd(&ad[c], xs[c]);
    }
}

// ---------------------------------------------------------------------------
// y = bf16((1+eps)*x + agg)
// ---------------------------------------------------------------------------
__global__ void gin_pre_kernel(const float* __restrict__ x,
                               const float* __restrict__ agg,
                               const float* __restrict__ eps, int layer,
                               u16* __restrict__ ybf) {
    size_t i = ((size_t)blockIdx.x * 256 + threadIdx.x) * 4;
    float e1 = 1.0f + eps[layer];
    float4 xv = *(const float4*)(x + i);
    float4 av = *(const float4*)(agg + i);
    u32 lo = (u32)f2bf(fmaf(xv.x, e1, av.x)) | ((u32)f2bf(fmaf(xv.y, e1, av.y)) << 16);
    u32 hi = (u32)f2bf(fmaf(xv.z, e1, av.z)) | ((u32)f2bf(fmaf(xv.w, e1, av.w)) << 16);
    uint2 o; o.x = lo; o.y = hi;
    *(uint2*)(ybf + i) = o;
}

// ---------------------------------------------------------------------------
// bf16 WMMA GEMM: C[M,512] = A[M,512] * B[512,512] + bias, optional leaky,
// optional fp32 out (accumulate or overwrite) and/or bf16 out.
// 256 thr (8 waves), tile 128x128, K-chunk 32. A/B staged via TDM
// (tensor_load_to_lds) double-buffered in LDS; wave0 issues + waits
// TENSORcnt, workgroup barrier publishes to all waves.
// __launch_bounds__(256,1): full VGPR budget -> no accumulator spills.
// ---------------------------------------------------------------------------
__global__ __launch_bounds__(256, 1)
void gemm_bf16_kernel(const u16* __restrict__ A,
                      const u16* __restrict__ Bp,
                      const float* __restrict__ bias,
                      float* __restrict__ outF,
                      u16* __restrict__ outBF,
                      int M, int leaky, int accum) {
    __shared__ __align__(16) u16 Ash[2][128 * 32];   // row-major [row][k], 8KB each
    __shared__ __align__(16) u16 Bsh[2][4096];       // frag-packed, 8KB each

    int tid  = threadIdx.x;
    int wave = tid >> 5;
    int lane = tid & 31;
    int row0 = blockIdx.x * 128;
    int col0 = blockIdx.y * 128;
    u32 rowsRem = (u32)(M - row0);                   // TDM zero-fills rows >= this

    v8f zero = {0.f, 0.f, 0.f, 0.f, 0.f, 0.f, 0.f, 0.f};
    v8f acc[8];
#pragma unroll
    for (int nt = 0; nt < 8; ++nt) acc[nt] = zero;

    u32 aOff0 = (u32)(size_t)&Ash[0][0];
    u32 bOff0 = (u32)(size_t)&Bsh[0][0];

    const u16* Abase = A + (size_t)row0 * DD;
    // Prologue: stage K-chunk 0 into buffer 0
    if (wave == 0) {
        tdm_load_2d(aOff0, Abase, 32, 128, DD, rowsRem, DD);
        tdm_load_2d(bOff0, Bp + ((size_t)(blockIdx.y * 8) * 32) * 16,
                    4096, 0, 4096, 1, 4096);
    }

    // Per-wave frag base addresses (buffer 0); buffer 1 = +8192 bytes.
    int fr  = lane & 15;
    int fkb = (lane >> 4) * 16;
    const u16* aFrag0 = &Ash[0][0] + (wave * 16 + fr) * 32 + fkb;
    const u16* bFrag0 = &Bsh[0][0] + lane * 16;

#pragma unroll 1
    for (int kt = 0; kt < 16; ++kt) {
        int cur = kt & 1;
        if (wave == 0) {
            if (kt + 1 < 16) {
                // Prefetch next K-chunk into the alternate buffer via TDM
                tdm_load_2d(aOff0 + (u32)((cur ^ 1) * 8192), Abase + (kt + 1) * 32,
                            32, 128, DD, rowsRem, DD);
                tdm_load_2d(bOff0 + (u32)((cur ^ 1) * 8192),
                            Bp + ((size_t)((kt + 1) * 32 + blockIdx.y * 8) * 32) * 16,
                            4096, 0, 4096, 1, 4096);
                __builtin_amdgcn_s_wait_tensorcnt(2);   // current buffer landed
            } else {
                __builtin_amdgcn_s_wait_tensorcnt(0);
            }
        }
        __syncthreads();

        size_t bufOff = (size_t)cur * 4096;          // u16 elements (8KB)

        // A fragment (contiguous k = 16*(l>>4) + s, matching packed B): lane l -> row l&15, K-halves by l>>4
        Frag af;
        {
            const u16* ap = aFrag0 + bufOff;
            af.q[0] = *(const uint4*)(ap);
            af.q[1] = *(const uint4*)(ap + 8);
        }
#pragma unroll
        for (int nt = 0; nt < 8; ++nt) {
            Frag bfm;
            const uint4* p = (const uint4*)(bFrag0 + bufOff + nt * 512);
            bfm.q[0] = p[0];
            bfm.q[1] = p[1];
            acc[nt] = __builtin_amdgcn_wmma_f32_16x16x32_bf16(
                false, af.v, false, bfm.v, (short)0, acc[nt], false, false);
        }
        __syncthreads();   // all reads done before TDM overwrites this buffer
    }

    // Epilogue: C layout lane=col (l&15), VGPR v -> row v + (l>>4)*8
    int cbase = lane & 15;
    int rhalf = (lane >> 4) * 8;
#pragma unroll
    for (int nt = 0; nt < 8; ++nt) {
        int gcol = col0 + nt * 16 + cbase;
        float bv = bias[gcol];
#pragma unroll
        for (int v = 0; v < 8; ++v) {
            int gr = row0 + wave * 16 + rhalf + v;
            if (gr < M) {
                float val = acc[nt][v] + bv;
                if (leaky) val = val > 0.f ? val : 0.01f * val;
                size_t idx = (size_t)gr * DD + gcol;
                if (outF) {
                    if (accum) outF[idx] += val;
                    else       outF[idx]  = val;
                }
                if (outBF) outBF[idx] = f2bf(val);
            }
        }
    }
}

// ---------------------------------------------------------------------------
extern "C" void kernel_launch(void* const* d_in, const int* in_sizes, int n_in,
                              void* d_out, int out_size, void* d_ws, size_t ws_size,
                              hipStream_t stream) {
    const float* x_in    = (const float*)d_in[0];
    const int*   edge    = (const int*)d_in[1];
    const int*   batch   = (const int*)d_in[2];
    const float* proj_w  = (const float*)d_in[3];
    const float* proj_b  = (const float*)d_in[4];
    const float* lin1_w  = (const float*)d_in[5];
    const float* lin1_b  = (const float*)d_in[6];
    const float* lin2_w  = (const float*)d_in[7];
    const float* lin2_b  = (const float*)d_in[8];
    const float* eps     = (const float*)d_in[9];
    const float* pool_w  = (const float*)d_in[10];
    const float* pool_b  = (const float*)d_in[11];

    float* part = (float*)d_out;                       // [64, 512]
    float* xout = (float*)d_out + (size_t)BB * DD;     // [65536, 512]

    char* ws = (char*)d_ws;
    size_t o = 0;
    float* agg    = (float*)(ws + o); o += (size_t)NN * DD * 4;     // 128 MB
    u16*   ybf    = (u16*)(ws + o);   o += (size_t)NN * DD * 2;     //  64 MB
    u16*   ybf2   = (u16*)(ws + o);   o += (size_t)NN * DD * 2;     //  64 MB
    u32*   keys   = (u32*)(ws + o);   o += (size_t)BB * DD * 4;
    u16*   smaxbf = (u16*)(ws + o);   o += (size_t)BB * DD * 2;
    u16*   packw  = (u16*)(ws + o);   o += (size_t)10 * DD * DD * 2;

    // One-time bf16 fragment packing of all 10 weight matrices
    pack_weights_kernel<<<640, 256, 0, stream>>>(proj_w, lin1_w, lin2_w, pool_w, packw);

    dim3 gemmBig(512, 4);
    dim3 gemmSmall(1, 4);

    // part = segmax(x) @ proj_w + proj_b
    hipMemsetAsync(keys, 0, (size_t)BB * DD * 4, stream);
    segmax_kernel<<<NN, 256, 0, stream>>>(x_in, batch, keys);
    decode_segmax_kernel<<<(BB * DD) / 256, 256, 0, stream>>>(keys, smaxbf);
    gemm_bf16_kernel<<<gemmSmall, 256, 0, stream>>>(smaxbf, packw, proj_b,
                                                    part, nullptr, BB, 0, 0);

    const float* xcur = x_in;
    for (int i = 0; i < LL; ++i) {
        // agg = segment_sum(x[src], dst)
        hipMemsetAsync(agg, 0, (size_t)NN * DD * 4, stream);
        scatter_add_kernel<<<EE / 8, 256, 0, stream>>>(xcur, edge, agg);
        // y = bf16((1+eps)*x + agg)
        gin_pre_kernel<<<(NN * DD) / 1024, 256, 0, stream>>>(xcur, agg, eps, i, ybf);
        // h = leaky(y @ lin1 + b1)  -> bf16 only (feeds next GEMM)
        gemm_bf16_kernel<<<gemmBig, 256, 0, stream>>>(
            ybf, packw + (size_t)(1 + i) * DD * DD, lin1_b + i * DD,
            nullptr, ybf2, NN, 1, 0);
        // x = leaky(h @ lin2 + b2)  -> fp32 out (final/next-layer x)
        gemm_bf16_kernel<<<gemmBig, 256, 0, stream>>>(
            ybf2, packw + (size_t)(4 + i) * DD * DD, lin2_b + i * DD,
            xout, nullptr, NN, 1, 0);
        // part += segmax(x) @ pool_w + pool_b
        hipMemsetAsync(keys, 0, (size_t)BB * DD * 4, stream);
        segmax_kernel<<<NN, 256, 0, stream>>>(xout, batch, keys);
        decode_segmax_kernel<<<(BB * DD) / 256, 256, 0, stream>>>(keys, smaxbf);
        gemm_bf16_kernel<<<gemmSmall, 256, 0, stream>>>(
            smaxbf, packw + (size_t)(7 + i) * DD * DD, pool_b + i * DD,
            part, nullptr, BB, 0, 1);
        xcur = xout;
    }
}
